// Reins_24386824306738
// MI455X (gfx1250) — matrix operand order repaired and checked by reading, not compiled, not run on hardware
//
#include <hip/hip_runtime.h>
#include <hip/hip_bf16.h>

typedef __attribute__((ext_vector_type(16))) __bf16 v16bf;
typedef __attribute__((ext_vector_type(8)))  float  v8f;

#define PI_F 3.14159265358979323846f

__device__ __forceinline__ unsigned short f32_to_bf16(float f) {
    union { __bf16 b; unsigned short s; } u;
    u.b = (__bf16)f;                       // native v_cvt_*_bf16_f32 (RNE)
    return u.s;
}

__device__ __forceinline__ v8f wmma_bf16(v16bf a, v16bf b, v8f c) {
    return __builtin_amdgcn_wmma_f32_16x16x32_bf16(false, a, false, b, (short)0, c, false, false);
}

// async copy of 16 bytes global -> LDS (ASYNCcnt tracked)
__device__ __forceinline__ void async_g2l_b128(unsigned lds_addr, const void* gaddr) {
    asm volatile("global_load_async_to_lds_b128 %0, %1, off"
                 :: "v"(lds_addr), "v"(gaddr) : "memory");
}
__device__ __forceinline__ void wait_async0() {
    asm volatile("s_wait_asynccnt 0" ::: "memory");
}

// ---------------------------------------------------------------------------
// K0: constants (DCT matrix, 1-mask, ifft cos/sin) + bf16 conversions
// ---------------------------------------------------------------------------
__global__ void k_init(const float* __restrict__ lt, const int* __restrict__ layer_p,
                       const float* __restrict__ wdf,
                       float* __restrict__ cD, float* __restrict__ cM1,
                       float* __restrict__ cCC, float* __restrict__ cSS,
                       unsigned short* __restrict__ tokb, unsigned short* __restrict__ wdfb)
{
    const int layer = layer_p[0];
    const float* tok = lt + (size_t)layer * 102400;  // 100*1024
    const int total = 4096 + 112 * 1024 + 1024 * 1024;
    for (int t = blockIdx.x * blockDim.x + threadIdx.x; t < total; t += gridDim.x * blockDim.x) {
        if (t < 4096) {
            int which = t >> 10, ij = t & 1023, i = ij >> 5, j = ij & 31;
            if (which == 0) {
                // D[k=i][n=j] = cos(pi*(2n+1)*k/64) * (k==0 ? sqrt(1/32) : sqrt(2/32))
                float v = cosf(PI_F * (2.0f * (float)j + 1.0f) * (float)i / 64.0f);
                float s = (i == 0) ? sqrtf(1.0f / 32.0f) : sqrtf(2.0f / 32.0f);
                cD[ij] = v * s;
            } else if (which == 1) {
                float xy = (float)(i > j ? i : j);
                float m;
                if (xy <= 3.0f)       m = 1.0f - xy / (3.0f + 1e-10f) * 0.95f;
                else if (xy <= 22.0f) m = 0.01f;
                else                  m = (xy - 22.0f) / 10.0f * 0.3f;
                cM1[ij] = 1.0f - m;
            } else if (which == 2) {
                cCC[ij] = cosf(2.0f * PI_F * (float)i * (float)j / 32.0f) * 0.17677669529663689f;
            } else {
                cSS[ij] = sinf(2.0f * PI_F * (float)i * (float)j / 32.0f) * 0.17677669529663689f;
            }
        } else if (t < 4096 + 112 * 1024) {
            int q = t - 4096; int m = q >> 10, c = q & 1023;
            float v = (m < 100) ? tok[m * 1024 + c] : 0.0f;   // pad N to 112
            tokb[q] = f32_to_bf16(v);
        } else {
            int q = t - 4096 - 112 * 1024;
            wdfb[q] = f32_to_bf16(wdf[q]);
        }
    }
}

// ---------------------------------------------------------------------------
// K2: t2f[m] = tokens[m+1] @ W_t2f^T + b_t2f, stored transposed bf16 [1024][128]
// ---------------------------------------------------------------------------
__global__ __launch_bounds__(128) void k_t2f(const float* __restrict__ lt,
                                             const int* __restrict__ layer_p,
                                             const float* __restrict__ wt2f,
                                             const float* __restrict__ bt2f,
                                             unsigned short* __restrict__ t2ft)
{
    __shared__ float sW[1024];
    const int c = blockIdx.x;
    const int m = threadIdx.x;
    for (int k = threadIdx.x; k < 1024; k += blockDim.x) sW[k] = wt2f[(size_t)c * 1024 + k];
    __syncthreads();
    if (m < 99) {
        const int layer = layer_p[0];
        const float* tokrow = lt + (size_t)layer * 102400 + (size_t)(m + 1) * 1024;
        float acc = bt2f[c];
        for (int k = 0; k < 1024; ++k) acc = fmaf(sW[k], tokrow[k], acc);
        t2ft[c * 128 + m] = f32_to_bf16(acc);
    } else {
        t2ft[c * 128 + m] = 0;                 // zero-pad K to 128
    }
}

// ---------------------------------------------------------------------------
// K1: per-row 2D DCT + (1-mask): one wave per 32x32 image.
//     Row staged into LDS with async global->LDS copies (ASYNCcnt).
// ---------------------------------------------------------------------------
__global__ __launch_bounds__(128) void k_dct(const float* __restrict__ feats,
                                             const float* __restrict__ cD,
                                             const float* __restrict__ cM1,
                                             float* __restrict__ cal)
{
    __shared__ float sD[32 * 33];
    __shared__ float sM1[1024];
    __shared__ float sX[4][1024];
    __shared__ float sT[4][32 * 33];
    const int tid = threadIdx.x, lane = tid & 31, wv = tid >> 5;
    for (int t = tid; t < 1024; t += 128) {
        sD[(t >> 5) * 33 + (t & 31)] = cD[t];
        sM1[t] = cM1[t];
    }
    const int r = blockIdx.x * 4 + wv;
    const float* gsrc = feats + 32768 + (size_t)r * 1024;
    const unsigned lbase = (unsigned)(size_t)&sX[wv][0];
    #pragma unroll
    for (int it = 0; it < 8; ++it)
        async_g2l_b128(lbase + (unsigned)((it * 32 + lane) * 16), gsrc + (it * 32 + lane) * 4);
    wait_async0();
    __syncthreads();
    float col[32];
    #pragma unroll
    for (int k = 0; k < 32; ++k) col[k] = sX[wv][k * 32 + lane];    // column `lane`
    for (int i = 0; i < 32; ++i) {                                  // T = D * X
        float acc = 0.0f;
        #pragma unroll
        for (int k = 0; k < 32; ++k) acc = fmaf(sD[i * 33 + k], col[k], acc);
        sT[wv][i * 33 + lane] = acc;
    }
    __syncthreads();
    float* dst = cal + (size_t)r * 1024;
    for (int i = 0; i < 32; ++i) {                                  // F = T * D^T
        float acc = 0.0f;
        #pragma unroll
        for (int l = 0; l < 32; ++l) acc = fmaf(sT[wv][i * 33 + l], sD[lane * 33 + l], acc);
        dst[i * 32 + lane] = acc * sM1[i * 32 + lane];
    }
}

// ---------------------------------------------------------------------------
// K3: attn = softmax(cal @ tokens^T / 32); write shifted probs as bf16 [r][128]
//     one wave per 16-row block; 7 N-tiles (padded 112) x 32 K-steps of WMMA
// ---------------------------------------------------------------------------
__global__ __launch_bounds__(128) void k_attn(const float* __restrict__ cal,
                                              const unsigned short* __restrict__ tokb,
                                              unsigned short* __restrict__ attnS)
{
    __shared__ float slog[4][16 * 112];
    const int tid = threadIdx.x, lane = tid & 31, wv = tid >> 5;
    const int half = lane >> 4, ln = lane & 15;
    const int rowBase = (blockIdx.x * 4 + wv) * 16;
    const int M = rowBase + ln;
    const int klo = half * 16;

    v8f z = {0.f, 0.f, 0.f, 0.f, 0.f, 0.f, 0.f, 0.f};
    v8f acc[7];
    #pragma unroll
    for (int t = 0; t < 7; ++t) acc[t] = z;

    for (int kb = 0; kb < 32; ++kb) {
        const int K0 = kb * 32;
        union { v16bf v; unsigned short s[16]; } A;
        const float* ap = cal + (size_t)M * 1024 + K0 + klo;   // cvt f32->bf16 on the fly
        #pragma unroll
        for (int e = 0; e < 8; ++e) A.s[e] = f32_to_bf16(ap[e]);
        #pragma unroll
        for (int e = 0; e < 8; ++e) A.s[8 + e] = f32_to_bf16(ap[8 + e]);
        #pragma unroll
        for (int t = 0; t < 7; ++t) {
            const int n = t * 16 + ln;
            const v16bf b = *reinterpret_cast<const v16bf*>(tokb + (size_t)n * 1024 + K0 + half * 16);
            acc[t] = wmma_bf16(A.v, b, acc[t]);
        }
    }
    #pragma unroll
    for (int t = 0; t < 7; ++t)
        #pragma unroll
        for (int rr = 0; rr < 8; ++rr)
            slog[wv][(rr + half * 8) * 112 + t * 16 + ln] = acc[t][rr];
    __syncthreads();
    if (lane < 16) {
        float* lp = &slog[wv][lane * 112];
        float mx = -3.4e38f;
        for (int m = 0; m < 100; ++m) { float v = lp[m] * 0.03125f; lp[m] = v; mx = fmaxf(mx, v); }
        float sum = 0.0f;
        for (int m = 0; m < 100; ++m) { float e = __expf(lp[m] - mx); lp[m] = e; sum += e; }
        const float inv = 1.0f / sum;
        unsigned short* op = attnS + (size_t)(rowBase + lane) * 128;
        for (int m = 0; m < 99; ++m) op[m] = f32_to_bf16(lp[m + 1] * inv);  // drop m=0, shift
        #pragma unroll
        for (int m = 99; m < 128; ++m) op[m] = 0;
    }
}

// ---------------------------------------------------------------------------
// K4: tmpb = bf16(cal + attnS @ t2fT^T). 16x256 tile/wave, K=128.
// ---------------------------------------------------------------------------
__global__ __launch_bounds__(256) void k_df(const unsigned short* __restrict__ attnS,
                                            const unsigned short* __restrict__ t2ft,
                                            const float* __restrict__ cal,
                                            unsigned short* __restrict__ tmpb)
{
    const int tid = threadIdx.x, lane = tid & 31, wv = tid >> 5;
    const int colBlock = blockIdx.x & 3;
    const int rowBlock = (blockIdx.x >> 2) * 8 + wv;
    const int half = lane >> 4, ln = lane & 15;
    const int rowBase = rowBlock * 16;
    const int M = rowBase + ln;
    const int klo = half * 16;
    const int Nbase = colBlock * 256;

    union AF { v16bf v; uint4 q[2]; } A[4];
    #pragma unroll
    for (int ks = 0; ks < 4; ++ks) {
        const unsigned short* ap = attnS + (size_t)M * 128 + ks * 32 + klo;
        A[ks].q[0] = *reinterpret_cast<const uint4*>(ap);
        A[ks].q[1] = *reinterpret_cast<const uint4*>(ap + 8);
    }
    v8f z = {0.f, 0.f, 0.f, 0.f, 0.f, 0.f, 0.f, 0.f};
    v8f acc[16];
    #pragma unroll
    for (int t = 0; t < 16; ++t) acc[t] = z;
    #pragma unroll
    for (int t = 0; t < 16; ++t) {
        const int n = Nbase + t * 16 + ln;
        #pragma unroll
        for (int ks = 0; ks < 4; ++ks) {
            const v16bf b = *reinterpret_cast<const v16bf*>(t2ft + (size_t)n * 128 + ks * 32 + half * 16);
            acc[t] = wmma_bf16(A[ks].v, b, acc[t]);
        }
    }
    #pragma unroll
    for (int t = 0; t < 16; ++t) {
        #pragma unroll
        for (int rr = 0; rr < 8; ++rr) {
            const int Mo = rowBase + rr + half * 8;
            const int No = Nbase + t * 16 + ln;
            const size_t idx = (size_t)Mo * 1024 + No;
            tmpb[idx] = f32_to_bf16(cal[idx] + acc[t][rr]);
        }
    }
}

// ---------------------------------------------------------------------------
// K5: y = tmp @ W_df^T + b_df  ->  d_out x-region. 16x256 tile/wave, K=1024.
//     All 8 waves of a block share the same 256-col W panel (L0 reuse).
// ---------------------------------------------------------------------------
__global__ __launch_bounds__(256) void k_gemm(const unsigned short* __restrict__ tmpb,
                                              const unsigned short* __restrict__ wdfb,
                                              const float* __restrict__ bdf,
                                              float* __restrict__ out)
{
    const int tid = threadIdx.x, lane = tid & 31, wv = tid >> 5;
    const int colBlock = blockIdx.x & 3;
    const int rowBlock = (blockIdx.x >> 2) * 8 + wv;
    const int half = lane >> 4, ln = lane & 15;
    const int rowBase = rowBlock * 16;
    const int M = rowBase + ln;
    const int klo = half * 16;
    const int Nbase = colBlock * 256;

    v8f z = {0.f, 0.f, 0.f, 0.f, 0.f, 0.f, 0.f, 0.f};
    v8f acc[16];
    #pragma unroll
    for (int t = 0; t < 16; ++t) acc[t] = z;

    #pragma unroll 1
    for (int kb = 0; kb < 32; ++kb) {
        const int K0 = kb * 32;
        const unsigned short* ap = tmpb + (size_t)M * 1024 + K0 + klo;
        __builtin_prefetch(ap + 128, 0, 1);
        union { v16bf v; uint4 q[2]; } A;
        A.q[0] = *reinterpret_cast<const uint4*>(ap);
        A.q[1] = *reinterpret_cast<const uint4*>(ap + 8);
        #pragma unroll
        for (int t = 0; t < 16; ++t) {
            const int n = Nbase + t * 16 + ln;
            const v16bf b = *reinterpret_cast<const v16bf*>(wdfb + (size_t)n * 1024 + K0 + half * 16);
            acc[t] = wmma_bf16(A.v, b, acc[t]);
        }
    }
    float* yb = out + 32768;
    #pragma unroll
    for (int t = 0; t < 16; ++t) {
        #pragma unroll
        for (int rr = 0; rr < 8; ++rr) {
            const int Mo = rowBase + rr + half * 8;
            const int No = Nbase + t * 16 + ln;
            yb[(size_t)Mo * 1024 + No] = acc[t][rr] + bdf[No];
        }
    }
}

// ---------------------------------------------------------------------------
// K6: out = clip(C*Y*C^T - S*Y*S^T, 0, 255); row = x + out*scale (in-place y)
//     Row staged into LDS with async global->LDS copies (ASYNCcnt).
// ---------------------------------------------------------------------------
__global__ __launch_bounds__(128) void k_ifft(const float* __restrict__ feats,
                                              const float* __restrict__ scale_p,
                                              const float* __restrict__ cCC,
                                              const float* __restrict__ cSS,
                                              float* __restrict__ out)
{
    __shared__ float sCC[32 * 33];
    __shared__ float sSS[32 * 33];
    __shared__ float sY[4][1024];
    __shared__ float sTc[4][32 * 33];
    __shared__ float sTs[4][32 * 33];
    const int tid = threadIdx.x, lane = tid & 31, wv = tid >> 5;
    for (int t = tid; t < 1024; t += 128) {
        sCC[(t >> 5) * 33 + (t & 31)] = cCC[t];
        sSS[(t >> 5) * 33 + (t & 31)] = cSS[t];
    }
    const int r = blockIdx.x * 4 + wv;
    float* row = out + 32768 + (size_t)r * 1024;
    const unsigned lbase = (unsigned)(size_t)&sY[wv][0];
    #pragma unroll
    for (int it = 0; it < 8; ++it)
        async_g2l_b128(lbase + (unsigned)((it * 32 + lane) * 16), row + (it * 32 + lane) * 4);
    wait_async0();
    __syncthreads();
    float col[32];
    #pragma unroll
    for (int k = 0; k < 32; ++k) col[k] = sY[wv][k * 32 + lane];
    for (int i = 0; i < 32; ++i) {
        float ac = 0.0f, as = 0.0f;
        #pragma unroll
        for (int k = 0; k < 32; ++k) {
            ac = fmaf(sCC[i * 33 + k], col[k], ac);
            as = fmaf(sSS[i * 33 + k], col[k], as);
        }
        sTc[wv][i * 33 + lane] = ac;
        sTs[wv][i * 33 + lane] = as;
    }
    __syncthreads();
    const float scale = scale_p[0];
    const float* xs = feats + 32768 + (size_t)r * 1024;
    for (int i = 0; i < 32; ++i) {
        float acc = 0.0f;
        #pragma unroll
        for (int l = 0; l < 32; ++l) {
            acc = fmaf(sTc[wv][i * 33 + l], sCC[lane * 33 + l], acc);
            acc = fmaf(-sTs[wv][i * 33 + l], sSS[lane * 33 + l], acc);
        }
        float v = fminf(fmaxf(acc, 0.0f), 255.0f);
        row[i * 32 + lane] = xs[i * 32 + lane] + v * scale;
    }
}

// ---------------------------------------------------------------------------
// K7: cls token passthrough
// ---------------------------------------------------------------------------
__global__ void k_cls(const float* __restrict__ feats, float* __restrict__ out)
{
    const int t = blockIdx.x * blockDim.x + threadIdx.x;
    if (t < 8192) ((float4*)out)[t] = ((const float4*)feats)[t];
}

// ---------------------------------------------------------------------------
extern "C" void kernel_launch(void* const* d_in, const int* in_sizes, int n_in,
                              void* d_out, int out_size, void* d_ws, size_t ws_size,
                              hipStream_t stream)
{
    const float* feats = (const float*)d_in[0];
    const int*   layer = (const int*)d_in[1];
    const float* lt    = (const float*)d_in[2];
    const float* scale = (const float*)d_in[3];
    const float* wt2f  = (const float*)d_in[4];
    const float* bt2f  = (const float*)d_in[5];
    const float* wdf   = (const float*)d_in[6];
    const float* bdf   = (const float*)d_in[7];
    float* out = (float*)d_out;

    char* ws = (char*)d_ws;
    float*          cD    = (float*)(ws + 0);
    float*          cM1   = (float*)(ws + 4096);
    float*          cCC   = (float*)(ws + 8192);
    float*          cSS   = (float*)(ws + 12288);
    unsigned short* tokb  = (unsigned short*)(ws + 16384);       // 112x1024 bf16
    unsigned short* t2ft  = (unsigned short*)(ws + 245760);      // 1024x128 bf16
    unsigned short* wdfb  = (unsigned short*)(ws + 507904);      // 1024x1024 bf16
    unsigned short* attnS = (unsigned short*)(ws + 2605056);     // 32768x128 bf16
    float*          cal   = (float*)(ws + 10993664);             // 32768x1024 f32
    unsigned short* tmpb  = (unsigned short*)(ws + 145211392);   // 32768x1024 bf16

    k_init<<<4560, 256, 0, stream>>>(lt, layer, wdf, cD, cM1, cCC, cSS, tokb, wdfb);
    k_t2f <<<1024, 128, 0, stream>>>(lt, layer, wt2f, bt2f, t2ft);
    k_dct <<<8192, 128, 0, stream>>>(feats, cD, cM1, cal);
    k_attn<<< 512, 128, 0, stream>>>(cal, tokb, attnS);
    k_df  <<<1024, 256, 0, stream>>>(attnS, t2ft, cal, tmpb);
    k_gemm<<<1024, 256, 0, stream>>>(tmpb, wdfb, bdf, out);
    k_ifft<<<8192, 128, 0, stream>>>(feats, scale, cCC, cSS, out);
    k_cls <<<  32, 256, 0, stream>>>(feats, out);
}
